// BidirectionalMambaAdaLNBlock_65163243815271
// MI455X (gfx1250) — matrix operand order repaired and checked by reading, not compiled, not run on hardware
//
#include <hip/hip_runtime.h>
#include <hip/hip_bf16.h>
#include <math.h>
#include <stdint.h>

// ---------------- model dims ----------------
#define BATCH   2
#define SEQL    2048
#define BL      (BATCH * SEQL)      // 4096 rows
#define DMODEL  512
#define DINNER  1024
#define DSTATE  16
#define DCONV   4
#define DTRANK  32
#define MLPHID  2048

// ---------------- WMMA GEMM -----------------
typedef __attribute__((ext_vector_type(16))) __bf16 v16bf;
typedef __attribute__((ext_vector_type(8)))  float  v8f;

#define WGSZ 256
#define TM 128
#define TN 128
#define TKK 64
#define LDK 72   // skew: row stride 144B (16B aligned) -> async b128 chunks stay aligned

// ---- async global->LDS support (gfx1250), guarded so absence falls back ----
#ifndef HAVE_ASYNC_LDS
#if defined(__has_builtin)
#if __has_builtin(__builtin_amdgcn_global_load_async_to_lds_b128)
#define HAVE_ASYNC_LDS 1
#endif
#endif
#endif
#ifndef HAVE_ASYNC_LDS
#define HAVE_ASYNC_LDS 0
#endif

#if HAVE_ASYNC_LDS
// builtin signature (from compiler diagnostic): param0 is
// 'int __attribute__((vector_size(16))) addrspace(1)*' -- match it exactly.
typedef int gvec4i __attribute__((vector_size(16)));
typedef __attribute__((address_space(1))) gvec4i as1_v4i;
typedef __attribute__((address_space(3))) gvec4i as3_v4i;
// build addrspace pointers via integers: low 32 bits of a generic LDS pointer
// are the LDS byte offset (flat aperture truncation)
#define TO_AS1(p) ((as1_v4i*)(unsigned long long)(uintptr_t)(p))
#define TO_AS3(p) ((as3_v4i*)(unsigned)(uintptr_t)(p))
#if __has_builtin(__builtin_amdgcn_s_wait_asynccnt)
#define WAIT_ASYNC0() __builtin_amdgcn_s_wait_asynccnt(0)
#else
#define WAIT_ASYNC0() asm volatile("s_wait_asynccnt 0x0" ::: "memory")
#endif
#endif

// WMMA inner product over one K-tile held in LDS (As: [m][k], Bs: [n][k])
__device__ __forceinline__ void wmma_tile_compute(
    const __bf16 (*__restrict__ Asb)[LDK], const __bf16 (*__restrict__ Bsb)[LDK],
    v8f acc[2][4], int wm, int wn, int half, int l16)
{
#pragma unroll
    for (int kk = 0; kk < TKK; kk += 32) {
        v16bf afrag[2], bfrag[4];
#pragma unroll
        for (int mi = 0; mi < 2; ++mi) {
            int m = wm * 32 + mi * 16 + l16;
#pragma unroll
            for (int v = 0; v < 8; ++v) {
                int kb = kk + ((v & 3) << 1) + ((v >> 2) << 4) + (half << 3);
                afrag[mi][2 * v]     = Asb[m][kb];
                afrag[mi][2 * v + 1] = Asb[m][kb + 1];
            }
        }
#pragma unroll
        for (int nj = 0; nj < 4; ++nj) {
            int n = wn * 64 + nj * 16 + l16;
#pragma unroll
            for (int v = 0; v < 8; ++v) {
                int kb = kk + ((v & 3) << 1) + ((v >> 2) << 4) + (half << 3);
                bfrag[nj][2 * v]     = Bsb[n][kb];
                bfrag[nj][2 * v + 1] = Bsb[n][kb + 1];
            }
        }
#pragma unroll
        for (int mi = 0; mi < 2; ++mi)
#pragma unroll
            for (int nj = 0; nj < 4; ++nj)
                acc[mi][nj] = __builtin_amdgcn_wmma_f32_16x16x32_bf16(
                    false, afrag[mi], false, bfrag[nj],
                    (short)0, acc[mi][nj], false, false);
    }
}

// C = act(A @ Bt^T + bias*scale) [+ Res]
// A: M x K bf16 (row-major, lda); Bt: N x K bf16 (row-major, ldbt = weights pre-transposed)
// ACT: 0 none, 1 exact GELU, 2 softplus.  ASYNC: async global->LDS double buffering
// (requires M%TM==0, N%TN==0, K%TKK==0).
template <int ACT, bool HAS_BIAS, bool HAS_RES, bool ASYNC>
__global__ __launch_bounds__(WGSZ) void gemm_bf16_wmma_kernel(
    const __bf16* __restrict__ A, const __bf16* __restrict__ Bt,
    const float* __restrict__ bias, const float* __restrict__ Res,
    float* __restrict__ C,
    int M, int N, int K, int lda, int ldbt, int ldc, float bias_scale)
{
    __shared__ __align__(16) __bf16 As[2][TM][LDK];
    __shared__ __align__(16) __bf16 Bs[2][TN][LDK];

    const int tid  = threadIdx.x;
    const int lane = tid & 31;         // wave32
    const int wave = tid >> 5;         // 0..7
    const int wm   = wave & 3;
    const int wn   = wave >> 2;
    const int half = lane >> 4;
    const int l16  = lane & 15;

    const int m0 = blockIdx.y * TM;
    const int n0 = blockIdx.x * TN;

    v8f acc[2][4];
#pragma unroll
    for (int mi = 0; mi < 2; ++mi)
#pragma unroll
        for (int nj = 0; nj < 4; ++nj)
#pragma unroll
            for (int i = 0; i < 8; ++i) acc[mi][nj][i] = 0.0f;

#if HAVE_ASYNC_LDS
    if constexpr (ASYNC) {
        // issue one tile (A: TM x TKK, Bt: TN x TKK) as per-lane 16B async copies
        auto issue_tile = [&](int buf, int k0) {
#pragma unroll
            for (int i = 0; i < (TM * TKK * 2) / (16 * WGSZ); ++i) {   // 4 chunks
                int c = i * WGSZ + tid;
                int r = c >> 3;                 // row (128B = 8 chunks per row)
                int o = (c & 7) * 8;            // element offset (8 bf16 = 16B)
                __builtin_amdgcn_global_load_async_to_lds_b128(
                    TO_AS1(A + (size_t)(m0 + r) * lda + k0 + o),
                    TO_AS3(&As[buf][r][o]), 0, 0);
            }
#pragma unroll
            for (int i = 0; i < (TN * TKK * 2) / (16 * WGSZ); ++i) {
                int c = i * WGSZ + tid;
                int r = c >> 3;
                int o = (c & 7) * 8;
                __builtin_amdgcn_global_load_async_to_lds_b128(
                    TO_AS1(Bt + (size_t)(n0 + r) * ldbt + k0 + o),
                    TO_AS3(&Bs[buf][r][o]), 0, 0);
            }
        };
        const int nt = K / TKK;
        issue_tile(0, 0);
        for (int t = 0; t < nt; ++t) {
            const int cur = t & 1;
            WAIT_ASYNC0();                       // this wave's copies of tile `cur` done
            __syncthreads();                     // whole tile present for all waves
            if (t + 1 < nt) issue_tile(1 - cur, (t + 1) * TKK);
            wmma_tile_compute(As[cur], Bs[cur], acc, wm, wn, half, l16);
            __syncthreads();                     // tile `cur` consumed; safe to overwrite
        }
    } else
#endif
    {
        for (int k0 = 0; k0 < K; k0 += TKK) {
#pragma unroll
            for (int i = 0; i < (TM * TKK) / WGSZ; ++i) {
                int idx = i * WGSZ + tid;
                int m = idx >> 6, k = idx & (TKK - 1);
                int gm = m0 + m, gk = k0 + k;
                As[0][m][k] = (gm < M && gk < K) ? A[(size_t)gm * lda + gk] : (__bf16)0.0f;
            }
#pragma unroll
            for (int i = 0; i < (TN * TKK) / WGSZ; ++i) {
                int idx = i * WGSZ + tid;
                int n = idx >> 6, k = idx & (TKK - 1);
                int gn = n0 + n, gk = k0 + k;
                Bs[0][n][k] = (gn < N && gk < K) ? Bt[(size_t)gn * ldbt + gk] : (__bf16)0.0f;
            }
            if (k0 + TKK < K) {                  // gfx1250 global_prefetch_b8
                int pm = m0 + (tid >> 1);
                if (pm < M) __builtin_prefetch(&A[(size_t)pm * lda + k0 + TKK], 0, 1);
                int pn = n0 + (tid >> 1);
                if (pn < N) __builtin_prefetch(&Bt[(size_t)pn * ldbt + k0 + TKK], 0, 1);
            }
            __syncthreads();
            wmma_tile_compute(As[0], Bs[0], acc, wm, wn, half, l16);
            __syncthreads();
        }
    }

    // epilogue: C VGPR i -> M = base + half*8 + i; lane%16 -> N
#pragma unroll
    for (int mi = 0; mi < 2; ++mi) {
#pragma unroll
        for (int nj = 0; nj < 4; ++nj) {
            int gn = n0 + wn * 64 + nj * 16 + l16;
            if (gn >= N) continue;
#pragma unroll
            for (int i = 0; i < 8; ++i) {
                int gm = m0 + wm * 32 + mi * 16 + half * 8 + i;
                if (gm >= M) continue;
                float v = acc[mi][nj][i];
                if (HAS_BIAS) v += bias[gn] * bias_scale;
                if (ACT == 1) v = 0.5f * v * (1.0f + erff(v * 0.70710678118654752f));
                else if (ACT == 2) v = (v > 20.0f) ? v : log1pf(expf(v));
                if (HAS_RES) v += Res[(size_t)gm * ldc + gn];
                C[(size_t)gm * ldc + gn] = v;
            }
        }
    }
}

// ---------------- cast kernels ----------------
// dst (M x K compact bf16) = src (M x srcld f32)
__global__ __launch_bounds__(256) void cast_bf16_strided_kernel(
    const float* __restrict__ src, int srcld, __bf16* __restrict__ dst,
    int K, long long total)
{
    long long i = (long long)blockIdx.x * 256 + threadIdx.x;
    if (i >= total) return;
    long long m = i / K;
    int k = (int)(i - m * K);
    dst[i] = (__bf16)src[m * srcld + k];
}

// dst (N x K bf16) = transpose(src (K x N f32))
__global__ __launch_bounds__(256) void transpose_cast_kernel(
    const float* __restrict__ src, __bf16* __restrict__ dst, int K, int N)
{
    long long i = (long long)blockIdx.x * 256 + threadIdx.x;
    if (i >= (long long)K * N) return;
    int n = (int)(i / K), k = (int)(i % K);
    dst[i] = (__bf16)src[(size_t)k * N + n];
}

// ---------------- elementwise / glue kernels ----------------
__device__ __forceinline__ float silu_f(float x) { return x / (1.0f + expf(-x)); }

__global__ __launch_bounds__(256) void ada_mod_kernel(
    const float* __restrict__ ctx, const float* __restrict__ W,
    const float* __restrict__ bias, float* __restrict__ mod)
{
    int i = blockIdx.x * 256 + threadIdx.x;
    if (i >= BATCH * 2 * DMODEL) return;
    int b = i / (2 * DMODEL), j = i % (2 * DMODEL);
    float acc = bias[j];
    for (int k = 0; k < DMODEL; ++k) {
        float c = ctx[b * DMODEL + k];
        acc += silu_f(c) * W[(size_t)k * (2 * DMODEL) + j];
    }
    mod[i] = acc;
}

__global__ __launch_bounds__(256) void norm_mod_kernel(
    const float* __restrict__ x, const float* __restrict__ g, const float* __restrict__ be,
    const float* __restrict__ mod, const unsigned char* __restrict__ mask,
    float* __restrict__ h, float* __restrict__ hrev)
{
    const int row = blockIdx.x;
    const int b = row / SEQL, t = row % SEQL;
    const int tid = threadIdx.x;
    __shared__ float s1[256], s2[256];
    float a0 = x[(size_t)row * DMODEL + tid];
    float a1 = x[(size_t)row * DMODEL + 256 + tid];
    s1[tid] = a0 + a1;
    s2[tid] = a0 * a0 + a1 * a1;
    __syncthreads();
    for (int off = 128; off > 0; off >>= 1) {
        if (tid < off) { s1[tid] += s1[tid + off]; s2[tid] += s2[tid + off]; }
        __syncthreads();
    }
    float mu  = s1[0] * (1.0f / DMODEL);
    float var = s2[0] * (1.0f / DMODEL) - mu * mu;
    float inv = rsqrtf(var + 1e-5f);
    float mfac = mask[row] ? 0.0f : 1.0f;
    const float* shift = mod + (size_t)b * (2 * DMODEL);
    const float* scale = shift + DMODEL;
    size_t rrow = (size_t)(b * SEQL + (SEQL - 1 - t)) * DMODEL;
    for (int j = tid; j < DMODEL; j += 256) {
        float xn = (x[(size_t)row * DMODEL + j] - mu) * inv * g[j] + be[j];
        float hv = (xn * (1.0f + scale[j]) + shift[j]) * mfac;
        h[(size_t)row * DMODEL + j] = hv;
        hrev[rrow + j] = hv;
    }
}

__global__ __launch_bounds__(256) void norm2_kernel(
    const float* __restrict__ x, const float* __restrict__ g, const float* __restrict__ be,
    float* __restrict__ out)
{
    const int row = blockIdx.x;
    const int tid = threadIdx.x;
    __shared__ float s1[256], s2[256];
    float a0 = x[(size_t)row * DMODEL + tid];
    float a1 = x[(size_t)row * DMODEL + 256 + tid];
    s1[tid] = a0 + a1;
    s2[tid] = a0 * a0 + a1 * a1;
    __syncthreads();
    for (int off = 128; off > 0; off >>= 1) {
        if (tid < off) { s1[tid] += s1[tid + off]; s2[tid] += s2[tid + off]; }
        __syncthreads();
    }
    float mu  = s1[0] * (1.0f / DMODEL);
    float var = s2[0] * (1.0f / DMODEL) - mu * mu;
    float inv = rsqrtf(var + 1e-5f);
    for (int j = tid; j < DMODEL; j += 256) {
        out[(size_t)row * DMODEL + j] =
            (x[(size_t)row * DMODEL + j] - mu) * inv * g[j] + be[j];
    }
}

__global__ __launch_bounds__(256) void conv_silu_kernel(
    const float* __restrict__ xz, const float* __restrict__ w,
    const float* __restrict__ cb, float* __restrict__ xconv)
{
    int i = blockIdx.x * 256 + threadIdx.x;
    if (i >= BL * DINNER) return;
    int c = i % DINNER;
    int row = i / DINNER;
    int b = row / SEQL, t = row % SEQL;
    float acc = cb[c];
#pragma unroll
    for (int j = 0; j < DCONV; ++j) {
        int tt = t - (DCONV - 1) + j;
        if (tt >= 0)
            acc += xz[(size_t)(b * SEQL + tt) * (2 * DINNER) + c] * w[c * DCONV + j];
    }
    xconv[(size_t)row * DINNER + c] = silu_f(acc);
}

__global__ __launch_bounds__(256) void ssm_scan_kernel(
    const float* __restrict__ delta, const float* __restrict__ proj,
    const float* __restrict__ xconv, const float* __restrict__ A_log,
    const float* __restrict__ Dp, float* __restrict__ ycat)
{
    int gid = blockIdx.x * 256 + threadIdx.x;
    if (gid >= BATCH * DINNER) return;
    int b = gid / DINNER, c = gid % DINNER;
    float Arow[DSTATE], hst[DSTATE];
#pragma unroll
    for (int s = 0; s < DSTATE; ++s) {
        Arow[s] = -expf(A_log[c * DSTATE + s]);
        hst[s]  = 0.0f;
    }
    float Dc = Dp[c];
    for (int t = 0; t < SEQL; ++t) {
        size_t row = (size_t)(b * SEQL + t);
        float dt = delta[row * DINNER + c];
        float u  = xconv[row * DINNER + c];
        const float* pr = proj + row * (DTRANK + 2 * DSTATE);
        float du = dt * u;
        float y = 0.0f;
#pragma unroll
        for (int s = 0; s < DSTATE; ++s) {
            float Bv = pr[DTRANK + s];
            float Cv = pr[DTRANK + DSTATE + s];
            hst[s] = expf(dt * Arow[s]) * hst[s] + du * Bv;
            y += hst[s] * Cv;
        }
        ycat[row * (2 * DINNER) + c] = y + u * Dc;
    }
}

__global__ __launch_bounds__(256) void silu_z_kernel(
    const float* __restrict__ xz, float* __restrict__ ycat)
{
    int i = blockIdx.x * 256 + threadIdx.x;
    if (i >= BL * DINNER) return;
    int c = i % DINNER;
    size_t row = (size_t)(i / DINNER);
    float z = xz[row * (2 * DINNER) + DINNER + c];
    ycat[row * (2 * DINNER) + DINNER + c] = silu_f(z);
}

__global__ __launch_bounds__(256) void flip_kernel(
    const float* __restrict__ src, float* __restrict__ dst, int dim)
{
    int i = blockIdx.x * 256 + threadIdx.x;
    if (i >= BL * DMODEL) return;
    int d = i % dim;
    int row = i / dim;
    int b = row / SEQL, t = row % SEQL;
    dst[(size_t)(b * SEQL + (SEQL - 1 - t)) * dim + d] = src[(size_t)row * dim + d];
}

__global__ __launch_bounds__(256) void concat_kernel(
    const float* __restrict__ hf, const float* __restrict__ hb, float* __restrict__ cat2)
{
    int i = blockIdx.x * 256 + threadIdx.x;
    if (i >= BL * DMODEL) return;
    int j = i % DMODEL;
    size_t row = (size_t)(i / DMODEL);
    cat2[row * (2 * DMODEL) + j]          = hf[row * DMODEL + j];
    cat2[row * (2 * DMODEL) + DMODEL + j] = hb[row * DMODEL + j];
}

__global__ __launch_bounds__(256) void mask_kernel(
    float* __restrict__ out, const unsigned char* __restrict__ mask)
{
    int i = blockIdx.x * 256 + threadIdx.x;
    if (i >= BL * DMODEL) return;
    int row = i / DMODEL;
    if (mask[row]) out[i] = 0.0f;
}

// ---------------- host orchestration ----------------
extern "C" void kernel_launch(void* const* d_in, const int* in_sizes, int n_in,
                              void* d_out, int out_size, void* d_ws, size_t ws_size,
                              hipStream_t stream) {
    (void)in_sizes; (void)n_in; (void)out_size; (void)ws_size;

    const float* x    = (const float*)d_in[0];
    const unsigned char* pmask = (const unsigned char*)d_in[1];
    const float* ctx  = (const float*)d_in[2];
    const float* n1g  = (const float*)d_in[3];
    const float* n1b  = (const float*)d_in[4];
    const float* adaW = (const float*)d_in[5];
    const float* adab = (const float*)d_in[6];
    const float* mix[2][9];   // in_W,out_W,conv_W,conv_b,x_W,dt_W,dt_b,A_log,D
    for (int d = 0; d < 2; ++d)
        for (int i = 0; i < 9; ++i)
            mix[d][i] = (const float*)d_in[7 + d * 9 + i];
    const float* fusW = (const float*)d_in[25];
    const float* n2g  = (const float*)d_in[26];
    const float* n2b  = (const float*)d_in[27];
    const float* w1   = (const float*)d_in[28];
    const float* b1   = (const float*)d_in[29];
    const float* w2   = (const float*)d_in[30];
    const float* b2   = (const float*)d_in[31];

    // ---- f32 arena ----
    float* ws = (float*)d_ws;
    float* h     = ws;                    // BL*512
    float* hrev  = ws + 2097152;          // BL*512
    float* mod   = ws + 4194304;          // 2048
    float* xz    = ws + 4196352;          // BL*2048
    float* xconv = ws + 12584960;         // BL*1024
    float* proj  = ws + 16779264;         // BL*64
    float* delta = ws + 17041408;         // BL*1024
    float* ycat  = ws + 21235712;         // BL*2048
    float* hf    = ws + 29624320;         // BL*512
    float* hbtmp = h;                     // aliases (non-overlapping lifetimes)
    float* hb    = ycat;
    float* cat2  = xconv;
    float* x1    = hrev;
    float* n2    = delta;
    float* mlph  = xz;
    float* out   = (float*)d_out;

    // ---- bf16 arena (after f32 arena; 16B aligned) ----
    __bf16* p = (__bf16*)(ws + 31721472);
    __bf16 *wtin[2], *wtout[2], *wtx[2], *wtdt[2];
    for (int d = 0; d < 2; ++d) {
        wtin[d]  = p; p += (size_t)DMODEL * 2 * DINNER;   // 2048 x 512 (N x K)
        wtout[d] = p; p += (size_t)2 * DINNER * DMODEL;   // 512 x 2048
        wtx[d]   = p; p += (size_t)DINNER * (DTRANK + 2 * DSTATE); // 64 x 1024
        wtdt[d]  = p; p += (size_t)DTRANK * DINNER;       // 1024 x 32
    }
    __bf16* wtfus = p; p += (size_t)2 * DMODEL * DMODEL;  // 512 x 1024
    __bf16* wtw1  = p; p += (size_t)DMODEL * MLPHID;      // 2048 x 512
    __bf16* wtw2  = p; p += (size_t)MLPHID * DMODEL;      // 512 x 2048
    __bf16* Abf   = p;                                     // BL x 2048 scratch

    dim3 blk(WGSZ);
    auto ggrid = [](int M, int N) { return dim3((N + TN - 1) / TN, (M + TM - 1) / TM); };
    auto cgrid = [](long long n) { return dim3((unsigned)((n + 255) / 256)); };
    const int NE = (BL * DINNER + 255) / 256;
    const int NH = (BL * DMODEL + 255) / 256;

    // 0) weights -> bf16 transposed (N x K)
    for (int d = 0; d < 2; ++d) {
        transpose_cast_kernel<<<cgrid((long long)DMODEL * 2 * DINNER), blk, 0, stream>>>(
            mix[d][0], wtin[d], DMODEL, 2 * DINNER);
        transpose_cast_kernel<<<cgrid((long long)2 * DINNER * DMODEL), blk, 0, stream>>>(
            mix[d][1], wtout[d], 2 * DINNER, DMODEL);
        transpose_cast_kernel<<<cgrid((long long)DINNER * (DTRANK + 2 * DSTATE)), blk, 0, stream>>>(
            mix[d][4], wtx[d], DINNER, DTRANK + 2 * DSTATE);
        transpose_cast_kernel<<<cgrid((long long)DTRANK * DINNER), blk, 0, stream>>>(
            mix[d][5], wtdt[d], DTRANK, DINNER);
    }
    transpose_cast_kernel<<<cgrid((long long)2 * DMODEL * DMODEL), blk, 0, stream>>>(
        fusW, wtfus, 2 * DMODEL, DMODEL);
    transpose_cast_kernel<<<cgrid((long long)DMODEL * MLPHID), blk, 0, stream>>>(
        w1, wtw1, DMODEL, MLPHID);
    transpose_cast_kernel<<<cgrid((long long)MLPHID * DMODEL), blk, 0, stream>>>(
        w2, wtw2, MLPHID, DMODEL);

    // 1) adaLN modulation + layernorm/modulate
    ada_mod_kernel<<<cgrid(BATCH * 2 * DMODEL), blk, 0, stream>>>(ctx, adaW, adab, mod);
    norm_mod_kernel<<<BL, blk, 0, stream>>>(x, n1g, n1b, mod, pmask, h, hrev);

    // 2) two SSM mixers
    for (int d = 0; d < 2; ++d) {
        const float* hin  = (d == 0) ? h  : hrev;
        float*       hout = (d == 0) ? hf : hbtmp;
        // in-proj (async): (BL x 512) @ W(512 x 2048)
        cast_bf16_strided_kernel<<<cgrid((long long)BL * DMODEL), blk, 0, stream>>>(
            hin, DMODEL, Abf, DMODEL, (long long)BL * DMODEL);
        gemm_bf16_wmma_kernel<0, false, false, true><<<ggrid(BL, 2 * DINNER), blk, 0, stream>>>(
            Abf, wtin[d], nullptr, nullptr, xz,
            BL, 2 * DINNER, DMODEL, DMODEL, DMODEL, 2 * DINNER, 0.0f);
        conv_silu_kernel<<<NE, blk, 0, stream>>>(xz, mix[d][2], mix[d][3], xconv);
        // x-proj (sync; N=64): (BL x 1024) @ W(1024 x 64)
        cast_bf16_strided_kernel<<<cgrid((long long)BL * DINNER), blk, 0, stream>>>(
            xconv, DINNER, Abf, DINNER, (long long)BL * DINNER);
        gemm_bf16_wmma_kernel<0, false, false, false><<<ggrid(BL, DTRANK + 2 * DSTATE), blk, 0, stream>>>(
            Abf, wtx[d], nullptr, nullptr, proj,
            BL, DTRANK + 2 * DSTATE, DINNER, DINNER, DINNER, DTRANK + 2 * DSTATE, 0.0f);
        // dt-proj + softplus(dt + 2*dt_b) (sync; K=32)
        cast_bf16_strided_kernel<<<cgrid((long long)BL * DTRANK), blk, 0, stream>>>(
            proj, DTRANK + 2 * DSTATE, Abf, DTRANK, (long long)BL * DTRANK);
        gemm_bf16_wmma_kernel<2, true, false, false><<<ggrid(BL, DINNER), blk, 0, stream>>>(
            Abf, wtdt[d], mix[d][6], nullptr, delta,
            BL, DINNER, DTRANK, DTRANK, DTRANK, DINNER, 2.0f);
        // selective scan + gate
        ssm_scan_kernel<<<cgrid(BATCH * DINNER), blk, 0, stream>>>(
            delta, proj, xconv, mix[d][7], mix[d][8], ycat);
        silu_z_kernel<<<NE, blk, 0, stream>>>(xz, ycat);
        // out-proj (async): (BL x 2048) @ W(2048 x 512)
        cast_bf16_strided_kernel<<<cgrid((long long)BL * 2 * DINNER), blk, 0, stream>>>(
            ycat, 2 * DINNER, Abf, 2 * DINNER, (long long)BL * 2 * DINNER);
        gemm_bf16_wmma_kernel<0, false, false, true><<<ggrid(BL, DMODEL), blk, 0, stream>>>(
            Abf, wtout[d], nullptr, nullptr, hout,
            BL, DMODEL, 2 * DINNER, 2 * DINNER, 2 * DINNER, DMODEL, 0.0f);
    }

    // 3) fuse + residual
    flip_kernel<<<NH, blk, 0, stream>>>(hbtmp, hb, DMODEL);
    concat_kernel<<<NH, blk, 0, stream>>>(hf, hb, cat2);
    cast_bf16_strided_kernel<<<cgrid((long long)BL * 2 * DMODEL), blk, 0, stream>>>(
        cat2, 2 * DMODEL, Abf, 2 * DMODEL, (long long)BL * 2 * DMODEL);
    gemm_bf16_wmma_kernel<0, false, true, true><<<ggrid(BL, DMODEL), blk, 0, stream>>>(
        Abf, wtfus, nullptr, x, x1,
        BL, DMODEL, 2 * DMODEL, 2 * DMODEL, 2 * DMODEL, DMODEL, 0.0f);

    // 4) MLP
    norm2_kernel<<<BL, blk, 0, stream>>>(x1, n2g, n2b, n2);
    cast_bf16_strided_kernel<<<cgrid((long long)BL * DMODEL), blk, 0, stream>>>(
        n2, DMODEL, Abf, DMODEL, (long long)BL * DMODEL);
    gemm_bf16_wmma_kernel<1, true, false, true><<<ggrid(BL, MLPHID), blk, 0, stream>>>(
        Abf, wtw1, b1, nullptr, mlph,
        BL, MLPHID, DMODEL, DMODEL, DMODEL, MLPHID, 1.0f);
    cast_bf16_strided_kernel<<<cgrid((long long)BL * MLPHID), blk, 0, stream>>>(
        mlph, MLPHID, Abf, MLPHID, (long long)BL * MLPHID);
    gemm_bf16_wmma_kernel<0, true, true, true><<<ggrid(BL, DMODEL), blk, 0, stream>>>(
        Abf, wtw2, b2, x1, out,
        BL, DMODEL, MLPHID, MLPHID, MLPHID, DMODEL, 1.0f);

    // 5) padding mask
    mask_kernel<<<NH, blk, 0, stream>>>(out, pmask);
}
